// MoEFusion_4140348473603
// MI455X (gfx1250) — matrix operand order repaired and checked by reading, not compiled, not run on hardware
//
#include <hip/hip_runtime.h>
#include <hip/hip_bf16.h>
#include <math.h>

typedef __attribute__((ext_vector_type(16))) _Float16 v16h;
typedef __attribute__((ext_vector_type(8)))  _Float16 v8h;
typedef __attribute__((ext_vector_type(8)))  float    v8f;
typedef int v4i_g __attribute__((vector_size(16)));   // matches builtin param type

union V16 { v16h v; v8h h[2]; };

// Async global->LDS path (gfx1250): gate on builtin availability so the host
// pass and any toolchain lacking the builtins falls back to the proven
// global_load -> ds_store staging.
#if defined(__has_builtin)
#if __has_builtin(__builtin_amdgcn_global_load_async_to_lds_b128) && \
    __has_builtin(__builtin_amdgcn_s_wait_asynccnt)
#define USE_ASYNC_LDS 1
#endif
#endif
#ifndef USE_ASYNC_LDS
#define USE_ASYNC_LDS 0
#endif

__device__ inline float wave_sum32(float v) {
#pragma unroll
  for (int off = 16; off > 0; off >>= 1) v += __shfl_xor(v, off, 32);
  return v;
}

// ---------------------------------------------------------------------------
// Tiled WMMA GEMM:  C[M,N] = A[M,K](f16) * Bt[N,K](f16)^T + bias
// Block tile 128x128, K-step 32, 8 waves, wave tile 64x32 (4x2 wmma tiles).
// Double-buffered LDS fed by GLOBAL_LOAD_ASYNC_TO_LDS_B128 when available.
// EPI: 0 = f16 out, 1 = relu->f16 out, 2 = f32 out: resid + alpha*(acc+bias)
// ---------------------------------------------------------------------------
#define BM 128
#define BN 128
#define BK 32
#define LDT 40   // padded LDS stride (f16); 40*2=80B, multiple of 16B

#if USE_ASYNC_LDS
__device__ inline void async_copy16(const _Float16* g, _Float16* l) {
  __builtin_amdgcn_global_load_async_to_lds_b128(
      (__attribute__((address_space(1))) v4i_g*)g,
      (__attribute__((address_space(3))) v4i_g*)l,
      0, 0);
}
#endif

template <int EPI>
__global__ __launch_bounds__(256) void k_gemm(
    const _Float16* __restrict__ A,   // M x K
    const _Float16* __restrict__ Bt,  // N x K (transposed weights)
    const float* __restrict__ bias,   // N
    void* __restrict__ Cout,
    int M, int N, int K, int ldc,
    const float* __restrict__ resid,  // EPI==2
    const float* __restrict__ alpha)  // EPI==2
{
  __shared__ __align__(16) _Float16 At[2][BM * LDT];
  __shared__ __align__(16) _Float16 Bts[2][BN * LDT];

  const int tid  = threadIdx.x;
  const int lane = tid & 31;
  const int wave = tid >> 5;
  const int bm0  = blockIdx.y * BM;
  const int bn0  = blockIdx.x * BN;
  const int wm0  = (wave >> 2) * 64;   // wave row origin within block
  const int wn0  = (wave & 3) * 32;    // wave col origin within block
  const int r    = lane & 15;
  const int hl   = lane >> 4;          // lane half (0/1)

  // staging coords: 512 chunks of 8 f16 per tile, 2 per thread
  const int c0 = tid,        c1 = tid + 256;
  const int r0 = c0 >> 2,    col0 = (c0 & 3) << 3;
  const int r1 = c1 >> 2,    col1 = (c1 & 3) << 3;

  v8f acc[4][2] = {};

  // fragment load + 8 WMMAs from one LDS buffer pair
  auto compute = [&](const _Float16* at, const _Float16* bt) {
    V16 af[4];
#pragma unroll
    for (int mi = 0; mi < 4; ++mi) {
      const _Float16* p = at + (wm0 + mi * 16 + r) * LDT + hl * 8;
      af[mi].h[0] = *(const v8h*)p;         // K = hl*8 .. +7
      af[mi].h[1] = *(const v8h*)(p + 16);  // K = 16+hl*8 .. +7
    }
    V16 bf[2];
#pragma unroll
    for (int ni = 0; ni < 2; ++ni) {
      const _Float16* p = bt + (wn0 + ni * 16 + r) * LDT + hl * 8;
      bf[ni].h[0] = *(const v8h*)p;         // K = hl*8 .. +7
      bf[ni].h[1] = *(const v8h*)(p + 16);  // K = 16+hl*8 .. +7
    }
#pragma unroll
    for (int mi = 0; mi < 4; ++mi)
#pragma unroll
      for (int ni = 0; ni < 2; ++ni)
        acc[mi][ni] = __builtin_amdgcn_wmma_f32_16x16x32_f16(
            false, af[mi].v, false, bf[ni].v, (short)0, acc[mi][ni],
            false, false);
  };

#if USE_ASYNC_LDS
  // ---- async double-buffered pipeline ----
  auto stage = [&](int buf, int kb) {
    async_copy16(A  + (size_t)(bm0 + r0) * K + kb + col0, At[buf]  + r0 * LDT + col0);
    async_copy16(A  + (size_t)(bm0 + r1) * K + kb + col1, At[buf]  + r1 * LDT + col1);
    async_copy16(Bt + (size_t)(bn0 + r0) * K + kb + col0, Bts[buf] + r0 * LDT + col0);
    async_copy16(Bt + (size_t)(bn0 + r1) * K + kb + col1, Bts[buf] + r1 * LDT + col1);
  };
  const int nsteps = K / BK;
  stage(0, 0);
  for (int i = 0; i < nsteps; ++i) {
    const int cur = i & 1;
    if (i + 1 < nsteps) {
      stage(cur ^ 1, (i + 1) * BK);
      // async loads complete in order: <=4 outstanding means tile i is in LDS
      __builtin_amdgcn_s_wait_asynccnt(4);
    } else {
      __builtin_amdgcn_s_wait_asynccnt(0);
    }
    __syncthreads();           // all waves' writes to buf[cur] visible
    compute(At[cur], Bts[cur]);
    __syncthreads();           // all reads done before buf[cur] is re-staged
  }
#else
  // ---- synchronous fallback staging ----
  for (int kb = 0; kb < K; kb += BK) {
    *(v8h*)(At[0]  + r0 * LDT + col0) = *(const v8h*)(A  + (size_t)(bm0 + r0) * K + kb + col0);
    *(v8h*)(At[0]  + r1 * LDT + col1) = *(const v8h*)(A  + (size_t)(bm0 + r1) * K + kb + col1);
    *(v8h*)(Bts[0] + r0 * LDT + col0) = *(const v8h*)(Bt + (size_t)(bn0 + r0) * K + kb + col0);
    *(v8h*)(Bts[0] + r1 * LDT + col1) = *(const v8h*)(Bt + (size_t)(bn0 + r1) * K + kb + col1);
    __syncthreads();
    compute(At[0], Bts[0]);
    __syncthreads();
  }
#endif

  // ---- epilogue: lane (hl,r), VGPR v -> row = 8*hl+v, col = r (per tile) ----
#pragma unroll
  for (int ni = 0; ni < 2; ++ni) {
    const int col = bn0 + wn0 + ni * 16 + r;
    const float bv = bias ? bias[col] : 0.f;
#pragma unroll
    for (int mi = 0; mi < 4; ++mi) {
      const int rowb = bm0 + wm0 + mi * 16 + hl * 8;
#pragma unroll
      for (int v = 0; v < 8; ++v) {
        float x = acc[mi][ni][v] + bv;
        size_t idx = (size_t)(rowb + v) * ldc + col;
        if (EPI == 0) {
          ((_Float16*)Cout)[idx] = (_Float16)x;
        } else if (EPI == 1) {
          ((_Float16*)Cout)[idx] = (_Float16)(x > 0.f ? x : 0.f);
        } else {
          ((float*)Cout)[idx] = resid[idx] + alpha[0] * x;
        }
      }
    }
  }
}

// ---------------------------------------------------------------------------
// LayerNorm f32 -> f16, one wave per token.
// ---------------------------------------------------------------------------
__global__ __launch_bounds__(256) void k_ln_f32_f16(
    const float* __restrict__ in, const float* __restrict__ g,
    const float* __restrict__ b, _Float16* __restrict__ out, int Dn)
{
  const int lane  = threadIdx.x & 31;
  const int token = blockIdx.x * 8 + (threadIdx.x >> 5);
  const float* row = in + (size_t)token * Dn;
  float s = 0.f, s2 = 0.f;
  for (int j = lane; j < Dn; j += 32) { float x = row[j]; s += x; s2 += x * x; }
  s = wave_sum32(s); s2 = wave_sum32(s2);
  float m   = s / (float)Dn;
  float var = s2 / (float)Dn - m * m;
  float inv = rsqrtf(var + 1e-5f);
  _Float16* orow = out + (size_t)token * Dn;
  for (int j = lane; j < Dn; j += 32)
    orow[j] = (_Float16)((row[j] - m) * inv * g[j] + b[j]);
}

// ---------------------------------------------------------------------------
// In-place LayerNorm (+optional relu) on f16 activations, one wave per token.
// ---------------------------------------------------------------------------
template <int RELU>
__global__ __launch_bounds__(256) void k_ln_f16(
    _Float16* __restrict__ x, const float* __restrict__ g,
    const float* __restrict__ b, int Dn)
{
  const int lane  = threadIdx.x & 31;
  const int token = blockIdx.x * 8 + (threadIdx.x >> 5);
  _Float16* row = x + (size_t)token * Dn;
  float s = 0.f, s2 = 0.f;
  for (int j = lane; j < Dn; j += 32) { float v = (float)row[j]; s += v; s2 += v * v; }
  s = wave_sum32(s); s2 = wave_sum32(s2);
  float m   = s / (float)Dn;
  float var = s2 / (float)Dn - m * m;
  float inv = rsqrtf(var + 1e-5f);
  for (int j = lane; j < Dn; j += 32) {
    float y = ((float)row[j] - m) * inv * g[j] + b[j];
    if (RELU) y = y > 0.f ? y : 0.f;
    row[j] = (_Float16)y;
  }
}

// ---------------------------------------------------------------------------
// Expert output LN (Dn=256) + weighted combine into comb.
// ---------------------------------------------------------------------------
__global__ __launch_bounds__(256) void k_ln3_combine(
    const _Float16* __restrict__ eo, const float* __restrict__ g,
    const float* __restrict__ b, const float* __restrict__ wfull,
    float* __restrict__ comb, int e, int first)
{
  const int lane  = threadIdx.x & 31;
  const int token = blockIdx.x * 8 + (threadIdx.x >> 5);
  const _Float16* row = eo + (size_t)token * 256;
  float s = 0.f, s2 = 0.f;
  for (int j = lane; j < 256; j += 32) { float v = (float)row[j]; s += v; s2 += v * v; }
  s = wave_sum32(s); s2 = wave_sum32(s2);
  float m   = s * (1.f / 256.f);
  float var = s2 * (1.f / 256.f) - m * m;
  float inv = rsqrtf(var + 1e-5f);
  float w   = wfull[token * 8 + e];
  float* crow = comb + (size_t)token * 256;
  for (int j = lane; j < 256; j += 32) {
    float y = ((float)row[j] - m) * inv * g[j] + b[j];
    crow[j] = first ? (w * y) : (crow[j] + w * y);
  }
}

// ---------------------------------------------------------------------------
// f32 -> f16 cast with independent row strides.
// ---------------------------------------------------------------------------
__global__ void k_cast_f32_f16(const float* __restrict__ in,
                               _Float16* __restrict__ out,
                               int Mrows, int Dcols, int ldi, int ldo)
{
  size_t idx = (size_t)blockIdx.x * blockDim.x + threadIdx.x;
  if (idx >= (size_t)Mrows * Dcols) return;
  int t = (int)(idx / Dcols), j = (int)(idx % Dcols);
  out[(size_t)t * ldo + j] = (_Float16)in[(size_t)t * ldi + j];
}

// ---------------------------------------------------------------------------
// W[K,N] f32 -> Wt[N,K] f16 (convert + transpose).
// ---------------------------------------------------------------------------
__global__ void k_convtr(const float* __restrict__ W, _Float16* __restrict__ Wt,
                         int K, int N)
{
  size_t idx = (size_t)blockIdx.x * blockDim.x + threadIdx.x;
  if (idx >= (size_t)K * N) return;
  int n = (int)(idx / K), k = (int)(idx % K);
  Wt[idx] = (_Float16)W[(size_t)k * N + n];
}

// ---------------------------------------------------------------------------
// Router logits: tiny GEMM (E=8), scalar per (token, expert).
// ---------------------------------------------------------------------------
__global__ void k_logits(const _Float16* __restrict__ g2,
                         const float* __restrict__ Wg3,
                         const float* __restrict__ bg3,
                         float* __restrict__ logits, int K)
{
  int idx = blockIdx.x * blockDim.x + threadIdx.x;
  if (idx >= 8192 * 8) return;
  int e = idx & 7, t = idx >> 3;
  const _Float16* row = g2 + (size_t)t * K;
  float s = bg3[e];
  for (int k = 0; k < K; ++k) s += (float)row[k] * Wg3[k * 8 + e];
  logits[idx] = s;
}

// ---------------------------------------------------------------------------
// Softmax + top-2 + renormalized gate weights (dense per-expert layout).
// ---------------------------------------------------------------------------
__global__ void k_router(const float* __restrict__ logits,
                         float* __restrict__ probs, float* __restrict__ wfull)
{
  int t = blockIdx.x * blockDim.x + threadIdx.x;
  if (t >= 8192) return;
  float p[8], mx = -1e30f;
#pragma unroll
  for (int e = 0; e < 8; ++e) { p[e] = logits[t * 8 + e]; mx = fmaxf(mx, p[e]); }
  float s = 0.f;
#pragma unroll
  for (int e = 0; e < 8; ++e) { p[e] = expf(p[e] - mx); s += p[e]; }
  float inv = 1.f / s;
#pragma unroll
  for (int e = 0; e < 8; ++e) { p[e] *= inv; probs[t * 8 + e] = p[e]; }
  int i1 = 0; float v1 = p[0];
#pragma unroll
  for (int e = 1; e < 8; ++e) if (p[e] > v1) { v1 = p[e]; i1 = e; }
  int i2 = -1; float v2 = -1.f;
#pragma unroll
  for (int e = 0; e < 8; ++e) if (e != i1 && p[e] > v2) { v2 = p[e]; i2 = e; }
  float wn = 1.f / (v1 + v2 + 1e-8f);
#pragma unroll
  for (int e = 0; e < 8; ++e) wfull[t * 8 + e] = 0.f;
  wfull[t * 8 + i1] = v1 * wn;
  wfull[t * 8 + i2] = v2 * wn;
}

// ---------------------------------------------------------------------------
// Load-balance loss: single block, deterministic reduction.
// ---------------------------------------------------------------------------
__global__ __launch_bounds__(256) void k_lb(const float* __restrict__ probs,
                                            float* __restrict__ out)
{
  __shared__ float red[256];
  float psum[8] = {}, csum[8] = {};
  for (int t = threadIdx.x; t < 8192; t += 256) {
    float p[8];
#pragma unroll
    for (int e = 0; e < 8; ++e) { p[e] = probs[t * 8 + e]; psum[e] += p[e]; }
    int i1 = 0; float v1 = p[0];
#pragma unroll
    for (int e = 1; e < 8; ++e) if (p[e] > v1) { v1 = p[e]; i1 = e; }
    int i2 = -1; float v2 = -1.f;
#pragma unroll
    for (int e = 0; e < 8; ++e) if (e != i1 && p[e] > v2) { v2 = p[e]; i2 = e; }
    csum[i1] += 1.f; csum[i2] += 1.f;
  }
  float tot[16];
  for (int i = 0; i < 16; ++i) {
    red[threadIdx.x] = (i < 8) ? psum[i] : csum[i - 8];
    __syncthreads();
    for (int st = 128; st > 0; st >>= 1) {
      if ((int)threadIdx.x < st) red[threadIdx.x] += red[threadIdx.x + st];
      __syncthreads();
    }
    tot[i] = red[0];
    __syncthreads();
  }
  if (threadIdx.x == 0) {
    float imp = 0.f, ent = 0.f;
    for (int e = 0; e < 8; ++e) {
      float P = tot[e] * (1.f / 8192.f);
      float f = tot[8 + e] * (1.f / 8192.f);
      imp += f * P;
      ent -= P * logf(P + 1e-8f);
    }
    imp *= 8.f;
    float maxent = logf(8.f);
    out[0] = (imp + (maxent - ent) / maxent) * 0.01f;
  }
}

// ---------------------------------------------------------------------------
extern "C" void kernel_launch(void* const* d_in, const int* in_sizes, int n_in,
                              void* d_out, int out_size, void* d_ws, size_t ws_size,
                              hipStream_t stream) {
  (void)in_sizes; (void)n_in; (void)out_size; (void)ws_size;

  const float* id_emb  = (const float*)d_in[0];
  const float* content = (const float*)d_in[1];
  const float* collab  = (const float*)d_in[2];
  const float* ln_ct_g = (const float*)d_in[3];
  const float* ln_ct_b = (const float*)d_in[4];
  const float* ln_cb_g = (const float*)d_in[5];
  const float* ln_cb_b = (const float*)d_in[6];
  const float* Wc  = (const float*)d_in[7];   const float* bc  = (const float*)d_in[8];
  const float* Wcb = (const float*)d_in[9];   const float* bcb = (const float*)d_in[10];
  const float* Wg1 = (const float*)d_in[11];  const float* bg1 = (const float*)d_in[12];
  const float* lng_g = (const float*)d_in[13]; const float* lng_b = (const float*)d_in[14];
  const float* Wg2 = (const float*)d_in[15];  const float* bg2 = (const float*)d_in[16];
  const float* Wg3 = (const float*)d_in[17];  const float* bg3 = (const float*)d_in[18];
  const float* We1 = (const float*)d_in[19];  const float* be1 = (const float*)d_in[20];
  const float* lne1_g = (const float*)d_in[21]; const float* lne1_b = (const float*)d_in[22];
  const float* We2 = (const float*)d_in[23];  const float* be2 = (const float*)d_in[24];
  const float* We3 = (const float*)d_in[25];  const float* be3 = (const float*)d_in[26];
  const float* lne3_g = (const float*)d_in[27]; const float* lne3_b = (const float*)d_in[28];
  const float* Wout = (const float*)d_in[29]; const float* bout = (const float*)d_in[30];
  const float* alpha = (const float*)d_in[31];

  constexpr int NT = 8192, D = 1024, CT = 768, CB = 64, ID3 = 3072;
  constexpr int GH = 1536, GH2 = 768, H = 1024, H2 = 512, H4 = 256, E = 8;

  char* wsb = (char*)d_ws;
  size_t off = 0;
  auto carve = [&](size_t bytes) -> void* {
    off = (off + 255) & ~(size_t)255;
    void* p = wsb + off;
    off += bytes;
    return p;
  };

  _Float16* x       = (_Float16*)carve((size_t)NT * ID3 * 2);
  _Float16* ct_ln   = (_Float16*)carve((size_t)NT * CT * 2);
  _Float16* cb_ln   = (_Float16*)carve((size_t)NT * CB * 2);
  _Float16* g1      = (_Float16*)carve((size_t)NT * GH * 2);
  _Float16* g2      = (_Float16*)carve((size_t)NT * GH2 * 2);
  float*    logits  = (float*)carve((size_t)NT * E * 4);
  float*    probs   = (float*)carve((size_t)NT * E * 4);
  float*    wfull   = (float*)carve((size_t)NT * E * 4);
  _Float16* h1      = (_Float16*)carve((size_t)NT * H * 2);
  _Float16* h2      = (_Float16*)carve((size_t)NT * H2 * 2);
  _Float16* eo      = (_Float16*)carve((size_t)NT * H4 * 2);
  float*    comb    = (float*)carve((size_t)NT * H4 * 4);
  _Float16* combf16 = (_Float16*)carve((size_t)NT * H4 * 2);
  _Float16* WcT   = (_Float16*)carve((size_t)D * CT * 2);
  _Float16* WcbT  = (_Float16*)carve((size_t)D * CB * 2);
  _Float16* Wg1T  = (_Float16*)carve((size_t)GH * ID3 * 2);
  _Float16* Wg2T  = (_Float16*)carve((size_t)GH2 * GH * 2);
  _Float16* We1T  = (_Float16*)carve((size_t)H * ID3 * 2);
  _Float16* We2T  = (_Float16*)carve((size_t)H2 * H * 2);
  _Float16* We3T  = (_Float16*)carve((size_t)H4 * H2 * 2);
  _Float16* WoutT = (_Float16*)carve((size_t)D * H4 * 2);

  auto cdiv = [](size_t a, size_t b) { return (unsigned)((a + b - 1) / b); };
  auto convtr = [&](const float* W, _Float16* Wt, int K, int N) {
    k_convtr<<<cdiv((size_t)K * N, 256), 256, 0, stream>>>(W, Wt, K, N);
  };

  // --- input LN + build x = concat(id, ct@Wc+bc, cb@Wcb+bcb) in f16 ---
  k_ln_f32_f16<<<NT / 8, 256, 0, stream>>>(content, ln_ct_g, ln_ct_b, ct_ln, CT);
  k_ln_f32_f16<<<NT / 8, 256, 0, stream>>>(collab, ln_cb_g, ln_cb_b, cb_ln, CB);
  k_cast_f32_f16<<<cdiv((size_t)NT * D, 256), 256, 0, stream>>>(id_emb, x, NT, D, D, ID3);

  convtr(Wc, WcT, CT, D);
  convtr(Wcb, WcbT, CB, D);
  convtr(Wg1, Wg1T, ID3, GH);
  convtr(Wg2, Wg2T, GH, GH2);
  convtr(Wout, WoutT, H4, D);

  k_gemm<0><<<dim3(D / BN, NT / BM), 256, 0, stream>>>(
      ct_ln, WcT, bc, (void*)(x + D), NT, D, CT, ID3, nullptr, nullptr);
  k_gemm<0><<<dim3(D / BN, NT / BM), 256, 0, stream>>>(
      cb_ln, WcbT, bcb, (void*)(x + 2 * D), NT, D, CB, ID3, nullptr, nullptr);

  // --- router MLP ---
  k_gemm<0><<<dim3(GH / BN, NT / BM), 256, 0, stream>>>(
      x, Wg1T, bg1, (void*)g1, NT, GH, ID3, GH, nullptr, nullptr);
  k_ln_f16<1><<<NT / 8, 256, 0, stream>>>(g1, lng_g, lng_b, GH);
  k_gemm<1><<<dim3(GH2 / BN, NT / BM), 256, 0, stream>>>(
      g1, Wg2T, bg2, (void*)g2, NT, GH2, GH, GH2, nullptr, nullptr);
  k_logits<<<cdiv((size_t)NT * E, 256), 256, 0, stream>>>(g2, Wg3, bg3, logits, GH2);
  k_router<<<cdiv((size_t)NT, 256), 256, 0, stream>>>(logits, probs, wfull);

  // --- experts (dense over E, LN3+gate folded into combine) ---
  for (int e = 0; e < E; ++e) {
    convtr(We1 + (size_t)e * ID3 * H, We1T, ID3, H);
    k_gemm<0><<<dim3(H / BN, NT / BM), 256, 0, stream>>>(
        x, We1T, be1 + (size_t)e * H, (void*)h1, NT, H, ID3, H, nullptr, nullptr);
    k_ln_f16<1><<<NT / 8, 256, 0, stream>>>(h1, lne1_g + (size_t)e * H,
                                            lne1_b + (size_t)e * H, H);
    convtr(We2 + (size_t)e * H * H2, We2T, H, H2);
    k_gemm<1><<<dim3(H2 / BN, NT / BM), 256, 0, stream>>>(
        h1, We2T, be2 + (size_t)e * H2, (void*)h2, NT, H2, H, H2, nullptr, nullptr);
    convtr(We3 + (size_t)e * H2 * H4, We3T, H2, H4);
    k_gemm<0><<<dim3(H4 / BN, NT / BM), 256, 0, stream>>>(
        h2, We3T, be3 + (size_t)e * H4, (void*)eo, NT, H4, H2, H4, nullptr, nullptr);
    k_ln3_combine<<<NT / 8, 256, 0, stream>>>(
        eo, lne3_g + (size_t)e * H4, lne3_b + (size_t)e * H4, wfull, comb, e,
        (e == 0) ? 1 : 0);
  }

  // --- output projection + alpha residual, straight into d_out (f32) ---
  k_cast_f32_f16<<<cdiv((size_t)NT * H4, 256), 256, 0, stream>>>(
      comb, combf16, NT, H4, H4, H4);
  k_gemm<2><<<dim3(D / BN, NT / BM), 256, 0, stream>>>(
      combf16, WoutT, bout, d_out, NT, D, H4, D, id_emb, alpha);

  // --- load balancing loss scalar ---
  k_lb<<<1, 256, 0, stream>>>(probs, ((float*)d_out) + (size_t)NT * D);
}
